// MSAPair2MSA_5755256176807
// MI455X (gfx1250) — compile-verified, matrix operand order checked by reading, not run on hardware
//
#include <hip/hip_runtime.h>
#include <math.h>

// ---------------- problem constants ----------------
#define NSQ 192          // number of sequences (N)
#define LLZ 384          // length (L)
#define DM  256          // D_MSA
#define DP  128          // D_PAIR
#define HH  8            // heads
#define DHD 32           // head dim
#define HD  256          // H*DH
#define MROW (NSQ*LLZ)   // 73728 rows of the MSA matrix
#define FFD 1024         // 4*D_MSA
#define SCALE_F 0.1767766953f
#define EPSV 1e-5f

typedef __attribute__((ext_vector_type(16))) __bf16 v16bf;
typedef __attribute__((ext_vector_type(8)))  float  v8f;
typedef __attribute__((ext_vector_type(4)))  unsigned int u32x4;

union FragU { u32x4 u[2]; v16bf v; };

__device__ __forceinline__ unsigned short f2bf(float f) {
  union { float f; unsigned u; } x; x.f = f;
  unsigned r = x.u + 0x7fffu + ((x.u >> 16) & 1u);
  return (unsigned short)(r >> 16);
}
__device__ __forceinline__ float bf2f(unsigned short h) {
  union { unsigned u; float f; } x; x.u = ((unsigned)h) << 16;
  return x.f;
}

// CDNA5 async global->LDS copy, 16B per lane, tracked by ASYNCcnt.
__device__ __forceinline__ void async_ld16(unsigned lds_off, const void* gp) {
  asm volatile("global_load_async_to_lds_b128 %0, %1, off"
               :: "v"(lds_off), "v"((unsigned long long)gp)
               : "memory");
}
__device__ __forceinline__ unsigned lds_addr(const void* p) {
  return (unsigned)(unsigned long long)p;   // generic LDS addr: low 32 bits = LDS offset
}

// Load a 16x32 bf16 WMMA fragment (A-style; B uses same pattern on an N-major
// "Bt" tile).  Per ISA layout: row = lane&15, khalf = lane>>4, elements 0..7
// are K = kOff + khalf*8 .. +7, elements 8..15 are K = kOff+16+khalf*8 .. +7.
// stride must be a multiple of 8 halfs so the 16B LDS loads stay aligned.
__device__ __forceinline__ v16bf lds_frag(const unsigned short* tile, int rowBase,
                                          int stride, int kOff, int lane) {
  int r  = rowBase + (lane & 15);
  int kh = (lane >> 4) * 8;
  const unsigned short* p = tile + r * stride + kOff + kh;
  FragU f;
  f.u[0] = *(const u32x4*)(p);
  f.u[1] = *(const u32x4*)(p + 16);
  return f.v;
}

// ---------------- generic batched bf16 WMMA GEMM ----------------
// C[b][m][n] = sum_k A[b][m][k] * Bt[b][n][k]  (+bias[n]) (+act / +residual)
// batch z decomposed as b1 = z/nb2, b2 = z%nb2, offsets b1*s?1 + b2*s?2.
// mode: 0 none, 1 relu, 2 sigmoid, 3 add residual
// K-loop is double-buffered with GLOBAL_LOAD_ASYNC_TO_LDS_B128 (ASYNCcnt).
__global__ __launch_bounds__(256) void bgemm_kernel(
    const unsigned short* __restrict__ A, long long lda, long long sA1, long long sA2,
    const unsigned short* __restrict__ B, long long ldb, long long sB1, long long sB2,
    float* __restrict__ C, unsigned short* __restrict__ Cbf,
    long long ldc, long long sC1, long long sC2,
    const float* __restrict__ res, long long ldr, long long sR1, long long sR2,
    const float* __restrict__ bias,
    int M, int N, int K, int nb2, int mode)
{
  __shared__ __align__(16) unsigned short As[2][64 * 40];
  __shared__ __align__(16) unsigned short Bs[2][64 * 40];
  __shared__ __align__(16) unsigned short dump[8];

  int z  = blockIdx.z;
  int b1 = z / nb2;
  int b2 = z % nb2;
  const unsigned short* Ab = A + b1 * sA1 + b2 * sA2;
  const unsigned short* Bb = B + b1 * sB1 + b2 * sB2;
  long long offC = b1 * sC1 + b2 * sC2;
  long long offR = b1 * sR1 + b2 * sR2;

  int m0 = blockIdx.y * 64, n0 = blockIdx.x * 64;
  int t  = threadIdx.x;
  int lr = t >> 2;          // 0..63: tile row loaded by this thread
  int lc = (t & 3) * 8;     // 0,8,16,24: K sub-chunk
  int w    = t >> 5;        // wave 0..7
  int lane = t & 31;
  int rt = w & 3;           // wave's 16-row tile
  int cp = (w >> 2) * 2;    // wave's 16-col tile pair base (0 or 2)

  bool mok = (m0 + lr) < M;
  bool nok = (n0 + lr) < N;
  long long arow = (long long)(mok ? (m0 + lr) : 0) * lda + lc;
  long long brow = (long long)(nok ? (n0 + lr) : 0) * ldb + lc;

  unsigned aoff0 = lds_addr(&As[0][lr * 40 + lc]);
  unsigned aoff1 = lds_addr(&As[1][lr * 40 + lc]);
  unsigned boff0 = lds_addr(&Bs[0][lr * 40 + lc]);
  unsigned boff1 = lds_addr(&Bs[1][lr * 40 + lc]);
  unsigned doff  = lds_addr(&dump[0]);
  if (!mok) { u32x4 zv = {}; *(u32x4*)&As[0][lr*40+lc] = zv; *(u32x4*)&As[1][lr*40+lc] = zv;
              aoff0 = doff; aoff1 = doff; }
  if (!nok) { u32x4 zv = {}; *(u32x4*)&Bs[0][lr*40+lc] = zv; *(u32x4*)&Bs[1][lr*40+lc] = zv;
              boff0 = doff; boff1 = doff; }
  __syncthreads();   // zero-filled guard rows visible before any compute

  int nk = K >> 5;
  // prologue: tile 0 -> buf0, tile 1 -> buf1 (uniform: 2 async ops per tile per wave)
  async_ld16(aoff0, Ab + arow);
  async_ld16(boff0, Bb + brow);
  if (nk > 1) {
    async_ld16(aoff1, Ab + arow + 32);
    async_ld16(boff1, Bb + brow + 32);
  }
  if (mode == 3) __builtin_prefetch(&res[offR + (long long)(m0 + lr) * 0 + n0], 0, 1);

  v8f acc0 = {}, acc1 = {};
  for (int i = 0; i < nk; ++i) {
    if (i + 1 < nk) asm volatile("s_wait_asynccnt 0x2" ::: "memory");
    else            asm volatile("s_wait_asynccnt 0x0" ::: "memory");
    __syncthreads();                 // all waves' tile-i copies are in LDS
    const unsigned short* Ac = As[i & 1];
    const unsigned short* Bc = Bs[i & 1];
    v16bf af  = lds_frag(Ac, rt * 16, 40, 0, lane);
    v16bf bf0 = lds_frag(Bc, cp * 16, 40, 0, lane);
    v16bf bf1 = lds_frag(Bc, (cp + 1) * 16, 40, 0, lane);
    acc0 = __builtin_amdgcn_wmma_f32_16x16x32_bf16(false, af, false, bf0, (short)0, acc0, false, false);
    acc1 = __builtin_amdgcn_wmma_f32_16x16x32_bf16(false, af, false, bf1, (short)0, acc1, false, false);
    __syncthreads();                 // buffer free before re-issuing into it
    if (i + 2 < nk) {
      long long k2 = (long long)(i + 2) * 32;
      async_ld16((i & 1) ? aoff1 : aoff0, Ab + arow + k2);
      async_ld16((i & 1) ? boff1 : boff0, Bb + brow + k2);
    }
  }

  int kh = lane >> 4, nl = lane & 15;
  for (int half = 0; half < 2; ++half) {
    v8f acc = half ? acc1 : acc0;
    int gn = n0 + (cp + half) * 16 + nl;
    if (gn >= N) continue;
    #pragma unroll
    for (int e = 0; e < 8; ++e) {
      int gm = m0 + rt * 16 + kh * 8 + e;
      if (gm >= M) continue;
      float v = acc[e];
      if (bias) v += bias[gn];
      if (mode == 1)      v = fmaxf(v, 0.f);
      else if (mode == 2) v = 1.f / (1.f + expf(-v));
      else if (mode == 3) v += res[offR + (long long)gm * ldr + gn];
      long long ci = offC + (long long)gm * ldc + gn;
      if (C)   C[ci]   = v;
      if (Cbf) Cbf[ci] = f2bf(v);
    }
  }
}

// ---------------- fused column attention (per i, h, 32-query chunk) ---------
// S = (SCALE*Q) K^T (K=32), softmax over 192, O = P V, all via WMMA + LDS.
// K tile is staged with async global->LDS copies.
__global__ __launch_bounds__(256) void col_attn_kernel(
    const unsigned short* __restrict__ qc, const unsigned short* __restrict__ kc,
    const unsigned short* __restrict__ vc, float* __restrict__ rout)
{
  __shared__ __align__(16) union {
    struct { unsigned short Qs[32 * 40]; unsigned short Ks[NSQ * 40]; } ab;
    unsigned short Pb[32 * 200];
  } u;
  __shared__ __align__(16) unsigned short Vt[DHD * NSQ];
  __shared__ __align__(16) float S[32 * NSQ];

  int t = threadIdx.x;
  int i = blockIdx.z, h = blockIdx.y;
  int qb = blockIdx.x * 32;
  const long long nstride = (long long)LLZ * HH * DHD;
  const long long base    = ((long long)i * HH + h) * DHD;

  // K rows -> LDS via async copies (uniform: 3 per thread)
  for (int c = t; c < NSQ * 4; c += 256) {
    int n = c >> 2, ck = (c & 3) * 8;
    async_ld16(lds_addr(&u.ab.Ks[n * 40 + ck]),
               &kc[(long long)n * nstride + base + ck]);
  }
  // Q rows, scaled by 1/sqrt(DH)
  for (int c = t; c < 32 * 4; c += 256) {
    int r = c >> 2, ck = (c & 3) * 8;
    const unsigned short* src = &qc[(long long)(qb + r) * nstride + base + ck];
    #pragma unroll
    for (int e = 0; e < 8; ++e)
      u.ab.Qs[r * 40 + ck + e] = f2bf(bf2f(src[e]) * SCALE_F);
  }
  // V transposed (d, n) for the second GEMM's Bt layout
  for (int c = t; c < DHD * NSQ; c += 256) {
    int d = c / NSQ, n = c % NSQ;
    Vt[d * NSQ + n] = vc[(long long)n * nstride + base + d];
  }
  asm volatile("s_wait_asynccnt 0x0" ::: "memory");
  __syncthreads();

  int w = t >> 5, lane = t & 31;
  int kh = lane >> 4, nl = lane & 15;

  // S = Q K^T : 2x12 tiles, 3 per wave, single K=32 wmma each
  #pragma unroll
  for (int tt = 0; tt < 3; ++tt) {
    int tile = w * 3 + tt;
    int qt = tile / 12, ct = tile % 12;
    v16bf af = lds_frag(u.ab.Qs, qt * 16, 40, 0, lane);
    v16bf bf = lds_frag(u.ab.Ks, ct * 16, 40, 0, lane);
    v8f acc = {};
    acc = __builtin_amdgcn_wmma_f32_16x16x32_bf16(false, af, false, bf, (short)0, acc, false, false);
    #pragma unroll
    for (int e = 0; e < 8; ++e)
      S[(qt * 16 + kh * 8 + e) * NSQ + ct * 16 + nl] = acc[e];
  }
  __syncthreads();

  // row softmax -> bf16 P (overlaps Q/K LDS, now dead)
  if (t < 32) {
    float mx = -1e30f;
    for (int j = 0; j < NSQ; ++j) mx = fmaxf(mx, S[t * NSQ + j]);
    float sum = 0.f;
    for (int j = 0; j < NSQ; ++j) { float e = expf(S[t * NSQ + j] - mx); S[t * NSQ + j] = e; sum += e; }
    float inv = 1.f / sum;
    for (int j = 0; j < NSQ; ++j) u.Pb[t * 200 + j] = f2bf(S[t * NSQ + j] * inv);
  }
  __syncthreads();

  // O = P V : 32x32 output, 2x2 tiles on waves 0..3, K = 192
  if (w < 4) {
    int qt = w >> 1, dt = w & 1;
    v8f acc = {};
    for (int kk = 0; kk < NSQ / 32; ++kk) {
      v16bf af = lds_frag(u.Pb, qt * 16, 200, kk * 32, lane);
      v16bf bf = lds_frag(Vt,  dt * 16, NSQ, kk * 32, lane);
      acc = __builtin_amdgcn_wmma_f32_16x16x32_bf16(false, af, false, bf, (short)0, acc, false, false);
    }
    #pragma unroll
    for (int e = 0; e < 8; ++e) {
      int m = qt * 16 + kh * 8 + e;
      int d = dt * 16 + nl;
      rout[(long long)(qb + m) * nstride + base + d] = acc[e];
    }
  }
}

// ---------------- LayerNorm (D == blockDim == 256), bf16 output -------------
__global__ __launch_bounds__(256) void ln_bf16_kernel(
    const float* __restrict__ x, const float* __restrict__ g, const float* __restrict__ b,
    unsigned short* __restrict__ out, int D)
{
  __shared__ float red[256];
  long long row = blockIdx.x;
  int t = threadIdx.x;
  float v = x[row * D + t];
  red[t] = v; __syncthreads();
  for (int s = 128; s > 0; s >>= 1) { if (t < s) red[t] += red[t + s]; __syncthreads(); }
  float mean = red[0] / D; __syncthreads();
  float d = v - mean;
  red[t] = d * d; __syncthreads();
  for (int s = 128; s > 0; s >>= 1) { if (t < s) red[t] += red[t + s]; __syncthreads(); }
  float var = red[0] / D;
  out[row * D + t] = f2bf(d * rsqrtf(var + EPSV) * g[t] + b[t]);
}

// ---------------- fused pair double-LN + bias matvec -------------------------
// PB[h][i][j] = sum_c LN2(LN1(pair[i,j,:]))[c] * r_bpair[c][h]
__global__ __launch_bounds__(128) void pair_bias_kernel(
    const float* __restrict__ pair, const float* npg, const float* npb,
    const float* rnpg, const float* rnpb, const float* __restrict__ wb,
    float* __restrict__ PB)
{
  __shared__ float red[DP];
  __shared__ float y[DP];
  long long row = blockIdx.x;          // i*L + j
  int t = threadIdx.x;
  float v = pair[row * DP + t];
  red[t] = v; __syncthreads();
  for (int s = 64; s > 0; s >>= 1) { if (t < s) red[t] += red[t + s]; __syncthreads(); }
  float mean = red[0] / DP; __syncthreads();
  float d = v - mean;
  red[t] = d * d; __syncthreads();
  for (int s = 64; s > 0; s >>= 1) { if (t < s) red[t] += red[t + s]; __syncthreads(); }
  float var = red[0] / DP; __syncthreads();
  float y1 = d * rsqrtf(var + EPSV) * npg[t] + npb[t];
  red[t] = y1; __syncthreads();
  for (int s = 64; s > 0; s >>= 1) { if (t < s) red[t] += red[t + s]; __syncthreads(); }
  mean = red[0] / DP; __syncthreads();
  d = y1 - mean;
  red[t] = d * d; __syncthreads();
  for (int s = 64; s > 0; s >>= 1) { if (t < s) red[t] += red[t + s]; __syncthreads(); }
  var = red[0] / DP; __syncthreads();
  y[t] = d * rsqrtf(var + EPSV) * rnpg[t] + rnpb[t];
  __syncthreads();
  for (int h = 0; h < HH; ++h) {
    red[t] = y[t] * wb[t * HH + h]; __syncthreads();
    for (int s = 64; s > 0; s >>= 1) { if (t < s) red[t] += red[t + s]; __syncthreads(); }
    if (t == 0) PB[(long long)h * LLZ * LLZ + row] = red[0];
    __syncthreads();
  }
}

// ---------------- sequence weight: dot(tq, tk) + softmax over s --------------
__global__ __launch_bounds__(256) void seqw_kernel(
    const float* __restrict__ TQ, const unsigned short* __restrict__ TK,
    float* __restrict__ SW)
{
  __shared__ float red[256];
  __shared__ float tqs[DHD];
  int qh = blockIdx.x;                 // q*H + h
  int q = qh / HH, h = qh % HH;
  int t = threadIdx.x;
  if (t < DHD) tqs[t] = TQ[(long long)q * HD + h * DHD + t] * SCALE_F;
  __syncthreads();
  bool act = t < NSQ;
  float val = 0.f;
  if (act) {
    const unsigned short* kp = &TK[((long long)t * LLZ + q) * HD + h * DHD];
    float s = 0.f;
    for (int d = 0; d < DHD; ++d) s += tqs[d] * bf2f(kp[d]);
    val = s;
  }
  red[t] = act ? val : -1e30f; __syncthreads();
  for (int s = 128; s > 0; s >>= 1) { if (t < s) red[t] = fmaxf(red[t], red[t + s]); __syncthreads(); }
  float mx = red[0]; __syncthreads();
  float e = act ? expf(val - mx) : 0.f;
  red[t] = e; __syncthreads();
  for (int s = 128; s > 0; s >>= 1) { if (t < s) red[t] += red[t + s]; __syncthreads(); }
  if (act) SW[(long long)qh * NSQ + t] = e / red[0];
}

// ---------------- row softmax over last dim, bf16 out ------------------------
__global__ __launch_bounds__(256) void softmax_bf16_kernel(
    const float* __restrict__ X, unsigned short* __restrict__ Y, int C)
{
  __shared__ float red[256];
  long long row = blockIdx.x;
  int t = threadIdx.x;
  const float* xr = X + row * C;
  float mx = -1e30f;
  for (int j = t; j < C; j += 256) mx = fmaxf(mx, xr[j]);
  red[t] = mx; __syncthreads();
  for (int s = 128; s > 0; s >>= 1) { if (t < s) red[t] = fmaxf(red[t], red[t + s]); __syncthreads(); }
  mx = red[0]; __syncthreads();
  float sum = 0.f;
  for (int j = t; j < C; j += 256) sum += expf(xr[j] - mx);
  red[t] = sum; __syncthreads();
  for (int s = 128; s > 0; s >>= 1) { if (t < s) red[t] += red[t + s]; __syncthreads(); }
  float inv = 1.f / red[0];
  for (int j = t; j < C; j += 256) Y[row * C + j] = f2bf(expf(xr[j] - mx) * inv);
}

// ---------------- small utility kernels --------------------------------------
__global__ void w2bf_t_kernel(const float* __restrict__ W, unsigned short* __restrict__ Wt,
                              int K, int N)
{
  long long idx = (long long)blockIdx.x * 256 + threadIdx.x;
  if (idx >= (long long)K * N) return;
  int n = (int)(idx % N); long long k = idx / N;
  Wt[(long long)n * K + k] = f2bf(W[idx]);
}

// qw[h][q][s*32+d] = qproj[s,q,h,d] * sw[q,h,s]
__global__ void repack_qw_kernel(const unsigned short* __restrict__ qproj,
                                 const float* __restrict__ SW,
                                 unsigned short* __restrict__ qw)
{
  long long idx = (long long)blockIdx.x * 256 + threadIdx.x;
  if (idx >= (long long)HH * LLZ * NSQ * DHD) return;
  int d = idx & 31; long long r = idx >> 5;
  int s = (int)(r % NSQ); r /= NSQ;
  int q = (int)(r % LLZ); int h = (int)(r / LLZ);
  float v = bf2f(qproj[(((long long)s * LLZ + q) * HH + h) * DHD + d]);
  qw[idx] = f2bf(v * SW[((long long)q * HH + h) * NSQ + s]);
}

// kt[h][k][s*32+d] = kproj[s,k,h,d] * SCALE
__global__ void repack_kt_kernel(const unsigned short* __restrict__ kproj,
                                 unsigned short* __restrict__ kt)
{
  long long idx = (long long)blockIdx.x * 256 + threadIdx.x;
  if (idx >= (long long)HH * LLZ * NSQ * DHD) return;
  int d = idx & 31; long long r = idx >> 5;
  int s = (int)(r % NSQ); r /= NSQ;
  int k = (int)(r % LLZ); int h = (int)(r / LLZ);
  kt[idx] = f2bf(bf2f(kproj[(((long long)s * LLZ + k) * HH + h) * DHD + d]) * SCALE_F);
}

// vt[s][h][d][k] = vproj[s,k,h,d]
__global__ void repack_vt_kernel(const unsigned short* __restrict__ vproj,
                                 unsigned short* __restrict__ vt)
{
  long long idx = (long long)blockIdx.x * 256 + threadIdx.x;
  if (idx >= (long long)NSQ * HH * DHD * LLZ) return;
  int k = (int)(idx % LLZ); long long r = idx / LLZ;
  int d = (int)(r % DHD); r /= DHD;
  int h = (int)(r % HH); int s = (int)(r / HH);
  vt[idx] = vproj[(((long long)s * LLZ + k) * HH + h) * DHD + d];
}

__global__ void gatemul_kernel(const float* __restrict__ gate, const float* __restrict__ out,
                               unsigned short* __restrict__ ga, long long total)
{
  long long idx = (long long)blockIdx.x * 256 + threadIdx.x;
  if (idx >= total) return;
  ga[idx] = f2bf(gate[idx] * out[idx]);
}

// ============================== host side ====================================
extern "C" void kernel_launch(void* const* d_in, const int* in_sizes, int n_in,
                              void* d_out, int out_size, void* d_ws, size_t ws_size,
                              hipStream_t stream) {
  (void)in_sizes; (void)n_in; (void)out_size; (void)ws_size;
  const float* msa_in  = (const float*)d_in[0];
  const float* pair_in = (const float*)d_in[1];
#define PIN(i) ((const float*)d_in[i])

  // ---- workspace layout ----
  char* base = (char*)d_ws;
  size_t off = 0;
  auto take = [&](size_t bytes) -> char* {
    char* p = base + off;
    off = (off + bytes + 255) & ~(size_t)255;
    return p;
  };
  float*          MS   = (float*)take((size_t)MROW * DM * 4);
  unsigned short* MLN  = (unsigned short*)take((size_t)MROW * DM * 2);
  float*          PB   = (float*)take((size_t)HH * LLZ * LLZ * 4);
  float*          TQ   = (float*)take((size_t)LLZ * HD * 4);
  unsigned short* TK   = (unsigned short*)take((size_t)MROW * HD * 2);
  float*          SW   = (float*)take((size_t)LLZ * HH * NSQ * 4);
  unsigned short* P0   = (unsigned short*)take((size_t)MROW * HD * 2);
  unsigned short* P1   = (unsigned short*)take((size_t)MROW * HD * 2);
  unsigned short* P2   = (unsigned short*)take((size_t)MROW * HD * 2);
  float*          GF   = (float*)take((size_t)MROW * HD * 4);      // gate (sigmoid)
  float*          ROUT = (float*)take((size_t)MROW * HD * 4);      // attn out
  unsigned short* FH   = (unsigned short*)GF;                      // FF hidden aliases GF+ROUT
  unsigned short* QW   = (unsigned short*)take((size_t)HH * LLZ * NSQ * DHD * 2);
  unsigned short* KT   = (unsigned short*)take((size_t)HH * LLZ * NSQ * DHD * 2);
  unsigned short* VT   = (unsigned short*)take((size_t)NSQ * HH * DHD * LLZ * 2);
  float*          LG   = (float*)take((size_t)HH * LLZ * LLZ * 4);
  unsigned short* ABF  = (unsigned short*)take((size_t)HH * LLZ * LLZ * 2);
  unsigned short* GA   = (unsigned short*)take((size_t)MROW * HD * 2);
  // bf16 transposed weights
  unsigned short* swq_t = (unsigned short*)take((size_t)DM * HD * 2);
  unsigned short* swk_t = (unsigned short*)take((size_t)DM * HD * 2);
  unsigned short* rq_t  = (unsigned short*)take((size_t)DM * HD * 2);
  unsigned short* rk_t  = (unsigned short*)take((size_t)DM * HD * 2);
  unsigned short* rv_t  = (unsigned short*)take((size_t)DM * HD * 2);
  unsigned short* rg_t  = (unsigned short*)take((size_t)DM * HD * 2);
  unsigned short* ro_t  = (unsigned short*)take((size_t)HD * DM * 2);
  unsigned short* cq_t  = (unsigned short*)take((size_t)DM * HD * 2);
  unsigned short* ck_t  = (unsigned short*)take((size_t)DM * HD * 2);
  unsigned short* cv_t  = (unsigned short*)take((size_t)DM * HD * 2);
  unsigned short* cg_t  = (unsigned short*)take((size_t)DM * HD * 2);
  unsigned short* co_t  = (unsigned short*)take((size_t)HD * DM * 2);
  unsigned short* f1_t  = (unsigned short*)take((size_t)DM * FFD * 2);
  unsigned short* f2_t  = (unsigned short*)take((size_t)FFD * DM * 2);

  auto bg = [&](const unsigned short* A, long long lda, long long sA1, long long sA2,
                const unsigned short* Bt, long long ldb, long long sB1, long long sB2,
                float* C, unsigned short* Cbf, long long ldc, long long sC1, long long sC2,
                const float* res, long long ldr, long long sR1, long long sR2,
                const float* bias, int M, int N, int K, int nb1, int nb2, int mode) {
    dim3 grid((N + 63) / 64, (M + 63) / 64, nb1 * nb2);
    bgemm_kernel<<<grid, 256, 0, stream>>>(A, lda, sA1, sA2, Bt, ldb, sB1, sB2,
                                           C, Cbf, ldc, sC1, sC2, res, ldr, sR1, sR2,
                                           bias, M, N, K, nb2, mode);
  };
  auto tw = [&](const float* W, unsigned short* Wt, int K, int N) {
    long long tot = (long long)K * N;
    w2bf_t_kernel<<<(unsigned)((tot + 255) / 256), 256, 0, stream>>>(W, Wt, K, N);
  };

  // ---- setup ----
  hipMemcpyAsync(MS, msa_in, (size_t)MROW * DM * 4, hipMemcpyDeviceToDevice, stream);
  tw(PIN(8),  swq_t, DM, HD);  tw(PIN(10), swk_t, DM, HD);
  tw(PIN(12), rq_t,  DM, HD);  tw(PIN(13), rk_t,  DM, HD);  tw(PIN(14), rv_t, DM, HD);
  tw(PIN(16), rg_t,  DM, HD);  tw(PIN(18), ro_t,  HD, DM);
  tw(PIN(22), cq_t,  DM, HD);  tw(PIN(23), ck_t,  DM, HD);  tw(PIN(24), cv_t, DM, HD);
  tw(PIN(25), cg_t,  DM, HD);  tw(PIN(27), co_t,  HD, DM);
  tw(PIN(31), f1_t,  DM, FFD); tw(PIN(33), f2_t,  FFD, DM);
  pair_bias_kernel<<<LLZ * LLZ, 128, 0, stream>>>(pair_in, PIN(2), PIN(3), PIN(6), PIN(7),
                                                  PIN(15), PB);

  const long long KSD = (long long)NSQ * DHD;     // 6144
  const long long tot = (long long)MROW * HD;     // 18.9M

  // ================= MSARowAttentionWithBias =================
  ln_bf16_kernel<<<MROW, 256, 0, stream>>>(MS, PIN(4), PIN(5), MLN, DM);
  // sequence-weight projections
  bg(MLN, DM,0,0, swq_t, DM,0,0, TQ, nullptr, HD,0,0, nullptr,0,0,0, PIN(9),  LLZ,  HD, DM, 1,1, 0);
  bg(MLN, DM,0,0, swk_t, DM,0,0, nullptr, TK, HD,0,0, nullptr,0,0,0, PIN(11), MROW, HD, DM, 1,1, 0);
  seqw_kernel<<<LLZ * HH, 256, 0, stream>>>(TQ, TK, SW);
  // q/k/v/gate projections
  bg(MLN, DM,0,0, rq_t, DM,0,0, nullptr, P0, HD,0,0, nullptr,0,0,0, nullptr, MROW, HD, DM, 1,1, 0);
  bg(MLN, DM,0,0, rk_t, DM,0,0, nullptr, P1, HD,0,0, nullptr,0,0,0, nullptr, MROW, HD, DM, 1,1, 0);
  bg(MLN, DM,0,0, rv_t, DM,0,0, nullptr, P2, HD,0,0, nullptr,0,0,0, nullptr, MROW, HD, DM, 1,1, 0);
  bg(MLN, DM,0,0, rg_t, DM,0,0, GF, nullptr, HD,0,0, nullptr,0,0,0, PIN(17), MROW, HD, DM, 1,1, 2);
  // repack for the big contractions
  repack_qw_kernel<<<(unsigned)((HH*(long long)LLZ*NSQ*DHD + 255)/256), 256, 0, stream>>>(P0, SW, QW);
  repack_kt_kernel<<<(unsigned)((HH*(long long)LLZ*NSQ*DHD + 255)/256), 256, 0, stream>>>(P1, KT);
  repack_vt_kernel<<<(unsigned)(((long long)NSQ*HH*DHD*LLZ + 255)/256), 256, 0, stream>>>(P2, VT);
  // logits: per head, (L x L), K = N*DH, bias fused as residual
  bg(QW, KSD, (long long)LLZ * KSD, 0,
     KT, KSD, (long long)LLZ * KSD, 0,
     LG, nullptr, LLZ, (long long)LLZ * LLZ, 0,
     PB, LLZ, (long long)LLZ * LLZ, 0,
     nullptr, LLZ, LLZ, (int)KSD, HH, 1, 3);
  softmax_bf16_kernel<<<HH * LLZ, 256, 0, stream>>>(LG, ABF, LLZ);
  // out[s,q,h,d] = attn[h] @ v[s,h] : batch (s,h)
  bg(ABF, LLZ, 0, (long long)LLZ * LLZ,
     VT,  LLZ, (long long)HH * DHD * LLZ, (long long)DHD * LLZ,
     ROUT, nullptr, HD, (long long)LLZ * HD, DHD,
     nullptr, 0,0,0, nullptr, LLZ, DHD, LLZ, NSQ, HH, 0);
  gatemul_kernel<<<(unsigned)((tot + 255)/256), 256, 0, stream>>>(GF, ROUT, GA, tot);
  bg(GA, HD,0,0, ro_t, HD,0,0, MS, nullptr, DM,0,0, MS, DM,0,0, PIN(19), MROW, DM, HD, 1,1, 3);

  // ================= MSAColAttention =================
  ln_bf16_kernel<<<MROW, 256, 0, stream>>>(MS, PIN(20), PIN(21), MLN, DM);
  bg(MLN, DM,0,0, cq_t, DM,0,0, nullptr, P0, HD,0,0, nullptr,0,0,0, nullptr, MROW, HD, DM, 1,1, 0);
  bg(MLN, DM,0,0, ck_t, DM,0,0, nullptr, P1, HD,0,0, nullptr,0,0,0, nullptr, MROW, HD, DM, 1,1, 0);
  bg(MLN, DM,0,0, cv_t, DM,0,0, nullptr, P2, HD,0,0, nullptr,0,0,0, nullptr, MROW, HD, DM, 1,1, 0);
  bg(MLN, DM,0,0, cg_t, DM,0,0, GF, nullptr, HD,0,0, nullptr,0,0,0, PIN(26), MROW, HD, DM, 1,1, 2);
  col_attn_kernel<<<dim3(NSQ / 32, HH, LLZ), 256, 0, stream>>>(P0, P1, P2, ROUT);
  gatemul_kernel<<<(unsigned)((tot + 255)/256), 256, 0, stream>>>(GF, ROUT, GA, tot);
  bg(GA, HD,0,0, co_t, HD,0,0, MS, nullptr, DM,0,0, MS, DM,0,0, PIN(28), MROW, DM, HD, 1,1, 3);

  // ================= FeedForward =================
  ln_bf16_kernel<<<MROW, 256, 0, stream>>>(MS, PIN(29), PIN(30), MLN, DM);
  bg(MLN, DM,0,0, f1_t, DM,0,0, nullptr, FH, FFD,0,0, nullptr,0,0,0, PIN(32), MROW, FFD, DM, 1,1, 1);
  bg(FH, FFD,0,0, f2_t, FFD,0,0, (float*)d_out, nullptr, DM,0,0, MS, DM,0,0, PIN(34), MROW, DM, FFD, 1,1, 3);
#undef PIN
}
